// HiddenLayer_704374636647
// MI455X (gfx1250) — matrix operand order repaired and checked by reading, not compiled, not run on hardware
//
#include <hip/hip_runtime.h>

// Problem constants (match reference)
#define Hh   256
#define Bb   256
#define Tt   2048
#define KC   8            // K chunks of 32 (H / 32)
#define LSTR 264          // padded LDS row stride in u16 (528B = 33*16B: aligned, bank-rotating)
#define NTW  2            // n-tiles per wave
#define NTHREADS 256
#define CHUNK 4096        // u16 elements per (b-block, t) fragment chunk: 16 rows * 256 cols

typedef __attribute__((ext_vector_type(16))) __bf16         v16bf;
typedef __attribute__((ext_vector_type(8)))  float          v8f;
typedef __attribute__((ext_vector_type(8)))  unsigned short u16x8;

union FragU {
    v16bf v;
    u16x8 h[2];
    unsigned short s[16];
};

// float -> bf16 (round-to-nearest-even), raw u16
static __device__ __forceinline__ unsigned short f2bf(float f) {
    unsigned int x = __builtin_bit_cast(unsigned int, f);
    x += 0x7fffu + ((x >> 16) & 1u);
    return (unsigned short)(x >> 16);
}
static __device__ __forceinline__ float bf2f(unsigned short h) {
    unsigned int x = ((unsigned int)h) << 16;
    return __builtin_bit_cast(float, x);
}

// tanh: single v_tanh_f32 trans op if the builtin exists on this toolchain,
// else exp-based (2 trans ops). Saturates correctly at +-1 either way.
static __device__ __forceinline__ float fast_tanh(float x) {
#if __has_builtin(__builtin_amdgcn_tanhf)
    return __builtin_amdgcn_tanhf(x);
#elif __has_builtin(__builtin_amdgcn_tanh_f32)
    return __builtin_amdgcn_tanh_f32(x);
#else
    const float e = __expf(2.0f * x);
    return 1.0f - 2.0f / (e + 1.0f);
#endif
}

// A fragment (16x32 bf16) from a padded LDS tile, row-major [row][k].
// Lanes 0-15: row=lane,  K = kb+0..7 (v0..3) and kb+16..23 (v4..7)
// Lanes 16-31: row=lane-16, K base shifted by 8.
static __device__ __forceinline__ v16bf lds_afrag(const unsigned short* base,
                                                  int row, int kc, int lane) {
    const int kb = kc * 32 + ((lane < 16) ? 0 : 8);
    const unsigned short* p = base + row * LSTR + kb;
    FragU u;
    u.h[0] = *(const u16x8*)(p);        // K kb..kb+7
    u.h[1] = *(const u16x8*)(p + 16);   // K kb+16..kb+23
    return u.v;
}

// B fragment (32x16 bf16) loaded once from row-major f32 W[k][n] in global
// memory; fragments then live in VGPRs for the whole kernel.
// Lane L<16: column n0+L, K = kb..kb+7 and kb+16..kb+23; lanes 16-31: K base +8.
static __device__ __forceinline__ v16bf g_bfrag(const float* __restrict__ W,
                                                int col, int kc, int lane) {
    const int kb = kc * 32 + ((lane < 16) ? 0 : 8);
    FragU u;
#pragma unroll
    for (int j = 0; j < 16; ++j)
        u.s[j] = f2bf(W[(size_t)(kb + (j < 8 ? j : j + 8)) * Hh + col]);
    return u.v;
}

// ---------------------------------------------------------------------------
// Phase 1: xw = bf16(x @ wx + bias), stored DIRECTLY in WMMA C-fragment order.
// Tiles are (16 batch rows at fixed t) == exactly the tiles phase 2 consumes.
// ws layout: chunk c = bblk*T + t (4096 u16 = 8 KiB); within a chunk:
//   offset = ntile*256 + lane*8  (each lane's 8 acc values packed, 16B).
// Bandwidth-bound => bf16 WMMA makes the math free; all stores are b128.
// ---------------------------------------------------------------------------
__global__ void __launch_bounds__(NTHREADS)
rnn_proj_kernel(const float* __restrict__ x, const float* __restrict__ wx,
                const float* __restrict__ bias, unsigned short* __restrict__ xw)
{
    __shared__ unsigned short xt[16 * LSTR];   // staged x tile, bf16, padded

    const int tid  = threadIdx.x;
    const int lane = tid & 31;
    const int wave = tid >> 5;
    const int mrow = lane & 15;

    // Per-wave weight fragments (pinned in VGPRs across all tiles)
    int   n0[NTW];
    float bval[NTW];
    v16bf bfrag[NTW][KC];
#pragma unroll
    for (int i = 0; i < NTW; ++i) {
        n0[i]   = (wave * NTW + i) * 16;
        bval[i] = bias[n0[i] + mrow];
#pragma unroll
        for (int kc = 0; kc < KC; ++kc)
            bfrag[i][kc] = g_bfrag(wx, n0[i] + mrow, kc, lane);
    }

    const int ntiles = (Bb / 16) * Tt;          // 32768 (bblk, t) tiles
    for (int mt = blockIdx.x; mt < ntiles; mt += gridDim.x) {
        const int bblk = mt >> 11;              // / Tt
        const int t    = mt & (Tt - 1);
        const int b0   = bblk * 16;

        __syncthreads();                        // previous tile fully consumed
        // stage 16 rows of x[(b0+k)*T + t, :] (1 KiB each, coalesced) as bf16
        for (int k = 0; k < 16; ++k)
            xt[k * LSTR + tid] = f2bf(x[((size_t)(b0 + k) * Tt + t) * Hh + tid]);
        __syncthreads();

        v8f acc[NTW];
#pragma unroll
        for (int i = 0; i < NTW; ++i)
#pragma unroll
            for (int j = 0; j < 8; ++j) acc[i][j] = bval[i];  // C = bias

#pragma unroll
        for (int kc = 0; kc < KC; ++kc) {
            const v16bf a = lds_afrag(xt, mrow, kc, lane);
#pragma unroll
            for (int i = 0; i < NTW; ++i)
                acc[i] = __builtin_amdgcn_wmma_f32_16x16x32_bf16(
                    false, a, false, bfrag[i][kc], (short)0, acc[i], false, false);
        }

        // pack fragment (8 x bf16 = 16B) and store coalesced b128
        const size_t cb = (size_t)mt * CHUNK;   // mt == bblk*T + t
#pragma unroll
        for (int i = 0; i < NTW; ++i) {
            u16x8 o;
#pragma unroll
            for (int j = 0; j < 8; ++j) o[j] = f2bf(acc[i][j]);
            *(u16x8*)(xw + cb + (size_t)(wave * NTW + i) * 256 + lane * 8) = o;
        }
    }
}

// ---------------------------------------------------------------------------
// Phase 2: sequential scan. One workgroup per 16 batch rows (16 independent
// chains), 8 waves splitting N. wh fragments pinned in VGPRs; h double-
// buffered in LDS (1 barrier/step). K split into two half-accumulators =>
// 4 independent WMMA streams of depth 4 (covers WMMA->WMMA RAW latency).
// Next step's C fragments preloaded into registers behind the WMMA chain.
// ---------------------------------------------------------------------------
__global__ void __launch_bounds__(NTHREADS)
rnn_scan_kernel(const float* __restrict__ wh,
                const unsigned short* __restrict__ xw,
                float* __restrict__ out)
{
    __shared__ unsigned short hbuf[2][16 * LSTR];  // h ping-pong, bf16, padded

    const int tid  = threadIdx.x;
    const int lane = tid & 31;
    const int wave = tid >> 5;
    const int mrow = lane & 15;
    const int mbase = (lane < 16) ? 0 : 8;
    const int bblk = blockIdx.x;

    // wh fragments: one-time global load -> VGPRs, reused 2048 times
    int   n0[NTW];
    v16bf bfrag[NTW][KC];
#pragma unroll
    for (int i = 0; i < NTW; ++i) {
        n0[i] = (wave * NTW + i) * 16;
#pragma unroll
        for (int kc = 0; kc < KC; ++kc)
            bfrag[i][kc] = g_bfrag(wh, n0[i] + mrow, kc, lane);
    }

    // h_0 = 0 in buffer 0
    for (int i = tid; i < 16 * LSTR; i += NTHREADS) hbuf[0][i] = 0;

    const size_t cbase = (size_t)bblk * Tt * CHUNK;  // this block's chunk run
    const size_t foff  = (size_t)(wave * NTW) * 256 + (size_t)lane * 8;

    // preload C fragments for t = 0
    u16x8 craw[NTW];
#pragma unroll
    for (int i = 0; i < NTW; ++i)
        craw[i] = *(const u16x8*)(xw + cbase + foff + (size_t)i * 256);

    __syncthreads();

    for (int t = 0; t < Tt; ++t) {
        const int rb = t & 1, wb = rb ^ 1;

        // two half-K accumulators per n-tile: accA = C + h@wh[K 0..127],
        // accB = h@wh[K 128..255]; merged after the chain.
        v8f accA[NTW], accB[NTW];
#pragma unroll
        for (int i = 0; i < NTW; ++i)
#pragma unroll
            for (int j = 0; j < 8; ++j) {
                accA[i][j] = bf2f(craw[i][j]);
                accB[i][j] = 0.0f;
            }

        // issue next step's C loads now: latency hides behind WMMA chain
        const int tn = (t + 1 < Tt) ? (t + 1) : t;
        u16x8 cnext[NTW];
#pragma unroll
        for (int i = 0; i < NTW; ++i)
            cnext[i] = *(const u16x8*)(xw + cbase + (size_t)tn * CHUNK + foff +
                                       (size_t)i * 256);
        // pull chunk t+6 toward L2/L0 (global_prefetch_b8)
        if (tid < 64) {
            const int tp = (t + 6 < Tt) ? (t + 6) : (Tt - 1);
            __builtin_prefetch(xw + cbase + (size_t)tp * CHUNK + (size_t)tid * 64, 0, 3);
        }

        // D = h_{t-1} @ wh + C : 4 independent WMMA streams, depth 4
#pragma unroll
        for (int kc = 0; kc < KC / 2; ++kc) {
            const v16bf a0 = lds_afrag(hbuf[rb], mrow, kc,          lane);
            const v16bf a1 = lds_afrag(hbuf[rb], mrow, kc + KC / 2, lane);
#pragma unroll
            for (int i = 0; i < NTW; ++i) {
                accA[i] = __builtin_amdgcn_wmma_f32_16x16x32_bf16(
                    false, a0, false, bfrag[i][kc], (short)0, accA[i], false, false);
                accB[i] = __builtin_amdgcn_wmma_f32_16x16x32_bf16(
                    false, a1, false, bfrag[i][kc + KC / 2], (short)0, accB[i], false, false);
            }
        }

        // h_t = tanh(accA + accB) -> write buffer, row-major for next A loads
#pragma unroll
        for (int i = 0; i < NTW; ++i) {
            const int col = n0[i] + mrow;
#pragma unroll
            for (int j = 0; j < 8; ++j) {
                const float v = accA[i][j] + accB[i][j];
                hbuf[wb][(mbase + j) * LSTR + col] = f2bf(fast_tanh(v));
            }
        }

#pragma unroll
        for (int i = 0; i < NTW; ++i) craw[i] = cnext[i];

        __syncthreads();   // single barrier per step (ping-pong buffers)
    }

    // T is even -> final h lives in hbuf[0]; write fp32, coalesced
    for (int i = tid; i < 16 * Hh; i += NTHREADS) {
        const int row = i >> 8, c = i & 255;
        out[(size_t)(bblk * 16 + row) * Hh + c] = bf2f(hbuf[0][row * LSTR + c]);
    }
}

// ---------------------------------------------------------------------------
extern "C" void kernel_launch(void* const* d_in, const int* in_sizes, int n_in,
                              void* d_out, int out_size, void* d_ws, size_t ws_size,
                              hipStream_t stream) {
    const float* x    = (const float*)d_in[0];   // [B, T, H]
    const float* wx   = (const float*)d_in[1];   // [H, H]
    const float* wh   = (const float*)d_in[2];   // [H, H]
    const float* bias = (const float*)d_in[3];   // [1, H]
    float* out = (float*)d_out;                  // [B, 1, H]
    unsigned short* xw = (unsigned short*)d_ws;  // fragment-ordered bf16, 256 MiB

    rnn_proj_kernel<<<1024, NTHREADS, 0, stream>>>(x, wx, bias, xw);
    rnn_scan_kernel<<<Bb / 16, NTHREADS, 0, stream>>>(wh, xw, out);
}
